// MultiHeadAttentionLayer_17703855194487
// MI455X (gfx1250) — matrix operand order repaired and checked by reading, not compiled, not run on hardware
//
#include <hip/hip_runtime.h>
#include <hip/hip_bf16.h>
#include <math.h>

// ---------------------------------------------------------------------------
// Graph-transformer attention layer for MI455X (gfx1250, wave32, WMMA).
//
// Shapes: N=50000 nodes, NE=800000 edges, IN=64, H=8 heads, D=16, HD=128.
// GEMMs on v_wmma_f32_16x16x32_f16; weights staged to LDS with the CDNA5
// async global->LDS path (ASYNCcnt), B fragments software-pipelined out of
// LDS so ds_load latency hides under the WMMA chain.
// ---------------------------------------------------------------------------

#define N_NODES   50000
#define N_EDGES   800000
#define IN_DIM    64
#define OUT_DIM   16
#define NUM_HEADS 8
#define HD        128   // NUM_HEADS * OUT_DIM
#define NT        8     // 16-wide N tiles across HD
#define KSTEPS_NODE 4   // K = 128 (concat h,p) / 32
#define KSTEPS_EDGE 2   // K = 64 / 32

typedef __attribute__((ext_vector_type(16))) _Float16 v16h;
typedef __attribute__((ext_vector_type(8)))  float    v8f;

// ---------------------------------------------------------------------------
// A-fragment loader: 16x32 f16 A tile in contiguous k order matching the packed B (not the ISA table order).
// Lane L holds row M = L&15; lanes 0-15 cover k-offsets {0..15},
// lanes 16-31 cover {16..31} -> one contiguous run of 16 per lane.
// ---------------------------------------------------------------------------
__device__ __forceinline__ v16h load_a_frag_f32(const float* __restrict__ rowptr,
                                                int kstep, int half) {
  const float* p = rowptr + kstep * 32 + half * 16;
  v16h a;
#pragma unroll
  for (int j = 0; j < 8; ++j) {
    a[j]     = (_Float16)p[j];
    a[8 + j] = (_Float16)p[8 + j];
  }
  return a;
}

// ---------------------------------------------------------------------------
// Async stage of `bytes` (multiple of nthreads*16) from global -> LDS using
// CDNA5 GLOBAL_LOAD_ASYNC_TO_LDS_B128 (tracked by ASYNCcnt). Each thread
// moves 16B per issue. Caller must __syncthreads() afterwards.
// ---------------------------------------------------------------------------
__device__ __forceinline__ void async_stage_to_lds(void* lds, const void* gptr,
                                                   int bytes, int tid, int nthreads) {
  uint32_t lds_off = (uint32_t)(uintptr_t)lds + (uint32_t)tid * 16u;
  uint64_t gaddr   = (uint64_t)(uintptr_t)gptr + (uint64_t)tid * 16u;
  const int stride = nthreads * 16;
  for (int off = 0; off < bytes; off += stride) {
    uint32_t l = lds_off + (uint32_t)off;
    uint64_t g = gaddr + (uint64_t)off;
    asm volatile("global_load_async_to_lds_b128 %0, %1, off"
                 :: "v"(l), "v"(g) : "memory");
  }
  asm volatile("s_wait_asynccnt 0x0" ::: "memory");
}

// ---------------------------------------------------------------------------
// Pack a row-major [Ktot x 128] fp32 weight into per-lane B-fragments:
// fragment(frag = nt*ksteps + ks): lane holds column n = nt*16 + (lane&15),
// k = ks*32 + (lane>>4)*16 + j (j=0..15), contiguous per lane -> 32B vector.
// ---------------------------------------------------------------------------
__global__ void pack_weight_kernel(const float* __restrict__ W,
                                   _Float16* __restrict__ out, int ksteps) {
  int t = blockIdx.x * blockDim.x + threadIdx.x;
  int total = NT * ksteps * 32 * 16;
  if (t >= total) return;
  int j    = t & 15;
  int lane = (t >> 4) & 31;
  int frag = t >> 9;
  int ks = frag % ksteps;
  int nt = frag / ksteps;
  int n = nt * 16 + (lane & 15);
  int k = ks * 32 + (lane >> 4) * 16 + j;
  out[t] = (_Float16)W[k * HD + n];
}

__global__ void zero_kernel(float* __restrict__ a, int n) {
  int i = blockIdx.x * blockDim.x + threadIdx.x;
  if (i < n) a[i] = 0.0f;
}

// ---------------------------------------------------------------------------
// One 16x128 GEMM tile for a single weight matrix, B from LDS.
// Software-pipelined: next N-tile's 4 B fragments are loaded from LDS before
// the current tile's WMMA chain so ds_load latency overlaps the matrix ops.
// Distinct pointer args per call keep stores in the global address space.
// ---------------------------------------------------------------------------
__device__ __forceinline__ void qkv_gemm_one(const _Float16* sW,
                                             const float* __restrict__ bias,
                                             float* __restrict__ out,
                                             const v16h* afrag, bool active,
                                             int m_base, int col, int half, int lane) {
  v16h b[KSTEPS_NODE];
#pragma unroll
  for (int ks = 0; ks < KSTEPS_NODE; ++ks)
    b[ks] = *(const v16h*)(sW + ((size_t)(0 * KSTEPS_NODE + ks) * 32 + lane) * 16);

#pragma unroll
  for (int nt = 0; nt < NT; ++nt) {
    v16h bn[KSTEPS_NODE];
    if (nt + 1 < NT) {
#pragma unroll
      for (int ks = 0; ks < KSTEPS_NODE; ++ks)
        bn[ks] = *(const v16h*)(sW + ((size_t)((nt + 1) * KSTEPS_NODE + ks) * 32 + lane) * 16);
    }
    v8f acc = {};
#pragma unroll
    for (int ks = 0; ks < KSTEPS_NODE; ++ks)
      acc = __builtin_amdgcn_wmma_f32_16x16x32_f16(
          /*neg_a=*/false, afrag[ks], /*neg_b=*/false, b[ks],
          /*c_mod=*/(short)0, acc, /*reuse_a=*/false, /*reuse_b=*/false);
    if (active) {
      const int n = nt * 16 + col;
      const float bv = bias[n];
#pragma unroll
      for (int r = 0; r < 8; ++r) {
        const int m = m_base + r + half * 8;
        out[(size_t)m * HD + n] = acc[r] + bv;
      }
    }
#pragma unroll
    for (int ks = 0; ks < KSTEPS_NODE; ++ks) b[ks] = bn[ks];
  }
}

// ---------------------------------------------------------------------------
// Node QKV: each wave computes a 16-row tile of Q, K, V (16x128 each).
// Weights staged one at a time through a 32KB LDS buffer (async path).
// ---------------------------------------------------------------------------
__global__ void __launch_bounds__(256)
node_qkv_kernel(const float* __restrict__ h, const float* __restrict__ p,
                const _Float16* __restrict__ Wq_p, const _Float16* __restrict__ Wk_p,
                const _Float16* __restrict__ Wv_p,
                const float* __restrict__ bq, const float* __restrict__ bk,
                const float* __restrict__ bv,
                float* __restrict__ Q, float* __restrict__ K, float* __restrict__ V) {
  __shared__ _Float16 sW[NT * KSTEPS_NODE * 32 * 16];   // 16384 halfs = 32 KB
  const int STAGE_BYTES = NT * KSTEPS_NODE * 32 * 16 * (int)sizeof(_Float16);

  const int tid  = threadIdx.x;
  const int lane = tid & 31;
  const int wave = tid >> 5;
  const int tile = blockIdx.x * 8 + wave;
  const bool active = (tile < N_NODES / 16);            // 3125 exact tiles
  const int tsafe  = active ? tile : 0;
  const int m_base = tsafe * 16;
  const int col  = lane & 15;
  const int half = lane >> 4;
  const int arow = m_base + col;

  v16h afrag[KSTEPS_NODE];
  afrag[0] = load_a_frag_f32(h + (size_t)arow * IN_DIM, 0, half);
  afrag[1] = load_a_frag_f32(h + (size_t)arow * IN_DIM, 1, half);
  afrag[2] = load_a_frag_f32(p + (size_t)arow * IN_DIM, 0, half);
  afrag[3] = load_a_frag_f32(p + (size_t)arow * IN_DIM, 1, half);

  // Q
  async_stage_to_lds(sW, Wq_p, STAGE_BYTES, tid, 256);
  __syncthreads();
  qkv_gemm_one(sW, bq, Q, afrag, active, m_base, col, half, lane);

  // K
  __syncthreads();
  async_stage_to_lds(sW, Wk_p, STAGE_BYTES, tid, 256);
  __syncthreads();
  qkv_gemm_one(sW, bk, K, afrag, active, m_base, col, half, lane);

  // V
  __syncthreads();
  async_stage_to_lds(sW, Wv_p, STAGE_BYTES, tid, 256);
  __syncthreads();
  qkv_gemm_one(sW, bv, V, afrag, active, m_base, col, half, lane);
}

// ---------------------------------------------------------------------------
// Fused edge kernel: each wave owns 16 edges.
//   E-tile [16x128] via WMMA (We staged to LDS once per block, async path),
//   per-head score = sum_D( K[src]*Q[dst]*E ) / sqrt(D) via 16-lane shfl tree,
//   score_soft = exp(clip(.,-5,5)) -> d_out,
//   scatter: atomicAdd wV[dst] += V[src]*score_soft, z[dst] += score_soft.
// Head h == N-tile h (OUT_DIM == 16): reductions come straight from the
// WMMA accumulator registers. Next head's B fragments are preloaded from LDS
// before the current head's WMMA chain (software pipelining).
// ---------------------------------------------------------------------------
__global__ void __launch_bounds__(128)
edge_attn_kernel(const float* __restrict__ e,
                 const int* __restrict__ src, const int* __restrict__ dst,
                 const _Float16* __restrict__ We_p, const float* __restrict__ be,
                 const float* __restrict__ Q, const float* __restrict__ K,
                 const float* __restrict__ V,
                 float* __restrict__ wV,    // [N, HD] accumulators (d_out)
                 float* __restrict__ z,     // [N, H]  accumulators (ws)
                 float* __restrict__ score) // [NE, H] (d_out)
{
  __shared__ _Float16 sWe[NT * KSTEPS_EDGE * 32 * 16];  // 8192 halfs = 16 KB

  const int tid   = threadIdx.x;
  const int lane  = tid & 31;
  const int wave  = tid >> 5;
  const int tile  = blockIdx.x * 4 + wave;  // 800000/16 = 50000 exact tiles
  const int ebase = tile * 16;
  const int col  = lane & 15;
  const int half = lane >> 4;
  const int arow = ebase + col;

  // stage We fragments into LDS (CDNA5 async global->LDS, ASYNCcnt tracked)
  async_stage_to_lds(sWe, We_p, NT * KSTEPS_EDGE * 32 * 16 * (int)sizeof(_Float16),
                     tid, 128);

  // speculative prefetch of the next tile's edge features (global_prefetch_b8)
  if (ebase + 64 < N_EDGES)
    __builtin_prefetch(e + (size_t)(ebase + 64 + col) * IN_DIM, 0, 0);

  v16h afrag[KSTEPS_EDGE];
  afrag[0] = load_a_frag_f32(e + (size_t)arow * IN_DIM, 0, half);
  afrag[1] = load_a_frag_f32(e + (size_t)arow * IN_DIM, 1, half);

  int sIdx[8], dIdx[8];
#pragma unroll
  for (int r = 0; r < 8; ++r) {
    const int eg = ebase + r + half * 8;
    sIdx[r] = src[eg];
    dIdx[r] = dst[eg];
  }

  __syncthreads();   // LDS staging visible to all waves

  const float scale = 0.25f;  // 1/sqrt(OUT_DIM)

  v16h b0 = *(const v16h*)(sWe + ((size_t)0 * 32 + lane) * 16);
  v16h b1 = *(const v16h*)(sWe + ((size_t)1 * 32 + lane) * 16);

#pragma unroll
  for (int head = 0; head < NUM_HEADS; ++head) {
    v16h n0, n1;
    if (head + 1 < NUM_HEADS) {
      n0 = *(const v16h*)(sWe + ((size_t)((head + 1) * KSTEPS_EDGE + 0) * 32 + lane) * 16);
      n1 = *(const v16h*)(sWe + ((size_t)((head + 1) * KSTEPS_EDGE + 1) * 32 + lane) * 16);
    }
    v8f acc = {};
    acc = __builtin_amdgcn_wmma_f32_16x16x32_f16(false, afrag[0], false, b0,
                                                 (short)0, acc, false, false);
    acc = __builtin_amdgcn_wmma_f32_16x16x32_f16(false, afrag[1], false, b1,
                                                 (short)0, acc, false, false);
    const int n = head * 16 + col;
    const float bn = be[n];
#pragma unroll
    for (int r = 0; r < 8; ++r) {
      const int eg = ebase + r + half * 8;
      const int s  = sIdx[r];
      const int d  = dIdx[r];
      const float Eval = acc[r] + bn;
      // per-lane contribution to score.sum(-1) for this (edge, head)
      float part = Eval * K[(size_t)s * HD + n] * Q[(size_t)d * HD + n] * scale;
      // reduce across the 16 lanes of this half-wave (xor masks stay in-group)
      part += __shfl_xor(part, 1);
      part += __shfl_xor(part, 2);
      part += __shfl_xor(part, 4);
      part += __shfl_xor(part, 8);
      const float sc    = fminf(fmaxf(part, -5.0f), 5.0f);
      const float ssoft = __expf(sc);
      if (col == 0) {
        score[(size_t)eg * NUM_HEADS + head] = ssoft;
        atomicAdd(&z[(size_t)d * NUM_HEADS + head], ssoft);
      }
      atomicAdd(&wV[(size_t)d * HD + n], V[(size_t)s * HD + n] * ssoft);
    }
    b0 = n0;
    b1 = n1;
  }
}

// h_out = wV / (z + 1e-6), in place over the d_out accumulator region.
__global__ void normalize_kernel(float* __restrict__ hout, const float* __restrict__ z) {
  int i = blockIdx.x * blockDim.x + threadIdx.x;
  if (i >= N_NODES * HD) return;
  int node = i >> 7;           // / 128
  int head = (i & 127) >> 4;   // (%128)/16
  hout[i] = hout[i] / (z[node * NUM_HEADS + head] + 1e-6f);
}

// ---------------------------------------------------------------------------
extern "C" void kernel_launch(void* const* d_in, const int* in_sizes, int n_in,
                              void* d_out, int out_size, void* d_ws, size_t ws_size,
                              hipStream_t stream) {
  const float* h   = (const float*)d_in[0];
  const float* p   = (const float*)d_in[1];
  const float* e   = (const float*)d_in[2];
  const int*   src = (const int*)d_in[3];
  const int*   dst = (const int*)d_in[4];
  const float* Wq  = (const float*)d_in[5];
  const float* bq  = (const float*)d_in[6];
  const float* Wk  = (const float*)d_in[7];
  const float* bk  = (const float*)d_in[8];
  const float* Wv  = (const float*)d_in[9];
  const float* bv  = (const float*)d_in[10];
  const float* We  = (const float*)d_in[11];
  const float* be  = (const float*)d_in[12];

  // ---- workspace layout ----
  char* ws = (char*)d_ws;
  size_t off = 0;
  const size_t packN = (size_t)NT * KSTEPS_NODE * 32 * 16;  // 16384 halfs
  const size_t packE = (size_t)NT * KSTEPS_EDGE * 32 * 16;  //  8192 halfs
  _Float16* Wq_p = (_Float16*)(ws + off); off += packN * sizeof(_Float16);
  _Float16* Wk_p = (_Float16*)(ws + off); off += packN * sizeof(_Float16);
  _Float16* Wv_p = (_Float16*)(ws + off); off += packN * sizeof(_Float16);
  _Float16* We_p = (_Float16*)(ws + off); off += packE * sizeof(_Float16);
  float* Q = (float*)(ws + off); off += (size_t)N_NODES * HD * sizeof(float);
  float* K = (float*)(ws + off); off += (size_t)N_NODES * HD * sizeof(float);
  float* V = (float*)(ws + off); off += (size_t)N_NODES * HD * sizeof(float);
  float* z = (float*)(ws + off); off += (size_t)N_NODES * NUM_HEADS * sizeof(float);

  // ---- outputs: h_out [N,HD] then score_soft [NE,H], concatenated flat ----
  float* hout  = (float*)d_out;
  float* score = hout + (size_t)N_NODES * HD;

  // 1) pack weights into WMMA B-fragment layout (tiny, ~112 KB total)
  pack_weight_kernel<<<(packN + 255) / 256, 256, 0, stream>>>(Wq, Wq_p, KSTEPS_NODE);
  pack_weight_kernel<<<(packN + 255) / 256, 256, 0, stream>>>(Wk, Wk_p, KSTEPS_NODE);
  pack_weight_kernel<<<(packN + 255) / 256, 256, 0, stream>>>(Wv, Wv_p, KSTEPS_NODE);
  pack_weight_kernel<<<(packE + 255) / 256, 256, 0, stream>>>(We, We_p, KSTEPS_EDGE);

  // 2) zero accumulators (h_out region of d_out, z in ws)
  zero_kernel<<<(N_NODES * HD + 255) / 256, 256, 0, stream>>>(hout, N_NODES * HD);
  zero_kernel<<<(N_NODES * NUM_HEADS + 255) / 256, 256, 0, stream>>>(z, N_NODES * NUM_HEADS);

  // 3) node QKV GEMMs: 3125 wave-tiles, 8 waves per 256-thread block
  node_qkv_kernel<<<(3125 + 7) / 8, 256, 0, stream>>>(h, p, Wq_p, Wk_p, Wv_p,
                                                      bq, bk, bv, Q, K, V);

  // 4) fused edge kernel: 50000 wave-tiles, 4 waves per 128-thread block
  edge_attn_kernel<<<50000 / 4, 128, 0, stream>>>(e, src, dst, We_p, be,
                                                  Q, K, V, hout, z, score);

  // 5) normalize
  normalize_kernel<<<(N_NODES * HD + 255) / 256, 256, 0, stream>>>(hout, z);
}
